// SpaceTimeTransformer_7413113553068
// MI455X (gfx1250) — matrix operand order repaired and checked by reading, not compiled, not run on hardware
//
#include <hip/hip_runtime.h>
#include <hip/hip_bf16.h>

typedef __attribute__((ext_vector_type(16))) _Float16 v16h;
typedef __attribute__((ext_vector_type(8)))  _Float16 v8h;
typedef __attribute__((ext_vector_type(8)))  float    v8f;

#define NTOK   16384   // tokens per stage view (B*S*H*W == B*H*W*S)
#define CDIM   192
#define NHEAD  6
#define MLPDIM 768

__device__ __forceinline__ float wave_sum(float v) {
  #pragma unroll
  for (int off = 16; off; off >>= 1) v += __shfl_xor(v, off, 32);
  return v;
}
__device__ __forceinline__ float wave_max(float v) {
  #pragma unroll
  for (int off = 16; off; off >>= 1) v = fmaxf(v, __shfl_xor(v, off, 32));
  return v;
}

// ---------------- weight pre-pack: f32 (K x N) row-major -> f16 WMMA B-fragment order
// out[((nt*KB + kb)*32 + lane)*16 + i] = W[kb*32 + 16*(lane>>4) + i][nt*16 + (lane&15)]
__global__ __launch_bounds__(256) void pack_w(const float* __restrict__ W,
                                              _Float16* __restrict__ out,
                                              int Kd, int Nd) {
  int op = blockIdx.x * 256 + threadIdx.x;
  if (op >= Kd * Nd) return;
  int i    = op & 15;
  int lane = (op >> 4) & 31;
  int rest = op >> 9;
  int KB   = Kd >> 5;
  int kb   = rest % KB;
  int nt   = rest / KB;
  int k = kb * 32 + ((lane >> 4) << 4) + i;
  int n = nt * 16 + (lane & 15);
  out[op] = (_Float16)W[(size_t)k * Nd + n];
}

// ---------------- LayerNorm (C=192), wave per token, writes f16 for next GEMM
__global__ __launch_bounds__(256) void ln_kernel(const float* __restrict__ x,
                                                 const float* __restrict__ g,
                                                 const float* __restrict__ b,
                                                 _Float16* __restrict__ y16, int N) {
  int wid = threadIdx.x >> 5, lane = threadIdx.x & 31;
  int token = blockIdx.x * 8 + wid;
  if (token >= N) return;
  const float* xr = x + (size_t)token * CDIM;
  float v[6]; float s = 0.f, s2 = 0.f;
  #pragma unroll
  for (int j = 0; j < 6; ++j) { v[j] = xr[lane + 32 * j]; s += v[j]; s2 += v[j] * v[j]; }
  s = wave_sum(s); s2 = wave_sum(s2);
  float m  = s * (1.f / 192.f);
  float rs = rsqrtf(s2 * (1.f / 192.f) - m * m + 1e-5f);
  _Float16* yr = y16 + (size_t)token * CDIM;
  #pragma unroll
  for (int j = 0; j < 6; ++j) {
    int c = lane + 32 * j;
    yr[c] = (_Float16)((v[j] - m) * rs * g[c] + b[c]);
  }
}

// ---------------- WMMA GEMM, 2x2 register blocking: one wave -> 32x32 output block
// out(N x Cout) = A(N x K, f16 row-major) * Bp(packed f16) + bias; optional GELU / residual
__global__ __launch_bounds__(256) void gemm16(const _Float16* __restrict__ A,
                                              const _Float16* __restrict__ Bp,
                                              const float* __restrict__ bias,
                                              const float* __restrict__ resid,
                                              float* __restrict__ outF,
                                              _Float16* __restrict__ outH,
                                              int N, int Kd, int Cout, int doGelu) {
  int wid = threadIdx.x >> 5, lane = threadIdx.x & 31;
  int npairs = Cout >> 5;                       // 32-col blocks
  int blk = blockIdx.x * 8 + wid;
  if (blk >= (N >> 5) * npairs) return;
  int tmp = blk / npairs, tnp = blk % npairs;   // 32x32 block coords
  int half_ = lane >> 4, lm = lane & 15;
  int KB = Kd >> 5;

  v8f c00 = {}, c01 = {}, c10 = {}, c11 = {};
  const _Float16* a0 = A + (size_t)(tmp * 32 + lm) * Kd + 16 * half_;
  const _Float16* a1 = a0 + (size_t)16 * Kd;
  const _Float16* b0 = Bp + (size_t)(2 * tnp) * KB * 512 + lane * 16;
  const _Float16* b1 = b0 + (size_t)KB * 512;
  for (int kb = 0; kb < KB; ++kb) {
    v16h fa0, fa1, fb0, fb1;
    *(v8h*)&fa0       = *(const v8h*)(a0);
    *((v8h*)&fa0 + 1) = *(const v8h*)(a0 + 8);
    *(v8h*)&fa1       = *(const v8h*)(a1);
    *((v8h*)&fa1 + 1) = *(const v8h*)(a1 + 8);
    fb0 = *(const v16h*)(b0);
    fb1 = *(const v16h*)(b1);
    c00 = __builtin_amdgcn_wmma_f32_16x16x32_f16(false, fa0, false, fb0, (short)0, c00, false, false);
    c01 = __builtin_amdgcn_wmma_f32_16x16x32_f16(false, fa0, false, fb1, (short)0, c01, false, false);
    c10 = __builtin_amdgcn_wmma_f32_16x16x32_f16(false, fa1, false, fb0, (short)0, c10, false, false);
    c11 = __builtin_amdgcn_wmma_f32_16x16x32_f16(false, fa1, false, fb1, (short)0, c11, false, false);
    a0 += 32; a1 += 32; b0 += 512; b1 += 512;
  }

  int gn0 = tnp * 32 + lm, gn1 = gn0 + 16;
  float bv0 = bias[gn0], bv1 = bias[gn1];
  #pragma unroll
  for (int r = 0; r < 8; ++r) {
    int gm0 = tmp * 32 + 8 * half_ + r;
    int gm1 = gm0 + 16;
    float v00 = c00[r] + bv0, v01 = c01[r] + bv1;
    float v10 = c10[r] + bv0, v11 = c11[r] + bv1;
    if (doGelu) {
      v00 = 0.5f * v00 * (1.f + erff(v00 * 0.70710678118654752f));
      v01 = 0.5f * v01 * (1.f + erff(v01 * 0.70710678118654752f));
      v10 = 0.5f * v10 * (1.f + erff(v10 * 0.70710678118654752f));
      v11 = 0.5f * v11 * (1.f + erff(v11 * 0.70710678118654752f));
    }
    size_t o00 = (size_t)gm0 * Cout + gn0, o01 = (size_t)gm0 * Cout + gn1;
    size_t o10 = (size_t)gm1 * Cout + gn0, o11 = (size_t)gm1 * Cout + gn1;
    if (resid) { v00 += resid[o00]; v01 += resid[o01]; v10 += resid[o10]; v11 += resid[o11]; }
    if (outF) { outF[o00] = v00; outF[o01] = v01; outF[o10] = v10; outF[o11] = v11; }
    if (outH) {
      outH[o00] = (_Float16)v00; outH[o01] = (_Float16)v01;
      outH[o10] = (_Float16)v10; outH[o11] = (_Float16)v11;
    }
  }
}

// ---------------- NA2D: 5x5 neighborhood attention over (16,32,32) grid, wave per (token, head)
__global__ __launch_bounds__(256) void na2d_kernel(const float* __restrict__ qkv,
                                                   const float* __restrict__ rpb,
                                                   _Float16* __restrict__ out) {
  int w = blockIdx.x * 8 + (threadIdx.x >> 5);
  if (w >= NTOK * NHEAD) return;
  int lane = threadIdx.x & 31;
  int head = w % NHEAD, token = w / NHEAD;
  int n = token >> 10, i = (token >> 5) & 31, j = token & 31;
  int si = i - 2; si = si < 0 ? 0 : (si > 27 ? 27 : si);
  int sj = j - 2; sj = sj < 0 ? 0 : (sj > 27 ? 27 : sj);
  const float* rp = rpb + head * 81;
  int hc = head * 32 + lane;

  float qv = qkv[(size_t)token * 576 + hc] * 0.17677669529663687f;
  float logit = -1e30f;
  #pragma unroll
  for (int p = 0; p < 25; ++p) {
    int a = p / 5, b = p % 5;
    int yi = si + a, xj = sj + b;
    int kt = (n << 10) + (yi << 5) + xj;
    float t = wave_sum(qv * qkv[(size_t)kt * 576 + 192 + hc]);
    t += rp[(yi - i + 4) * 9 + (xj - j + 4)];
    if (lane == p) logit = t;
  }
  float mx  = wave_max(logit);
  float e   = (lane < 25) ? __expf(logit - mx) : 0.f;
  float den = wave_sum(e);
  float wgt = e / den;

  float acc = 0.f;
  #pragma unroll
  for (int p = 0; p < 25; ++p) {
    int a = p / 5, b = p % 5;
    int kt = (n << 10) + ((si + a) << 5) + (sj + b);
    acc += __shfl(wgt, p, 32) * qkv[(size_t)kt * 576 + 384 + hc];
  }
  out[(size_t)token * CDIM + hc] = (_Float16)acc;
}

// ---------------- NA1D: K=5 over L=8 sequences, rows ordered (b*1024+p, s)
__global__ __launch_bounds__(256) void na1d_kernel(const float* __restrict__ qkv,
                                                   const float* __restrict__ rpb,
                                                   _Float16* __restrict__ out) {
  int w = blockIdx.x * 8 + (threadIdx.x >> 5);
  if (w >= NTOK * NHEAD) return;
  int lane = threadIdx.x & 31;
  int head = w % NHEAD, token = w / NHEAD;
  int bp = token >> 3, s = token & 7;
  int st = s - 2; st = st < 0 ? 0 : (st > 3 ? 3 : st);
  const float* rp = rpb + head * 9;
  int hc = head * 32 + lane;

  float qv = qkv[(size_t)token * 576 + hc] * 0.17677669529663687f;
  float logit = -1e30f;
  #pragma unroll
  for (int p = 0; p < 5; ++p) {
    int kt = (bp << 3) + st + p;
    float t = wave_sum(qv * qkv[(size_t)kt * 576 + 192 + hc]);
    t += rp[st + p - s + 4];
    if (lane == p) logit = t;
  }
  float mx  = wave_max(logit);
  float e   = (lane < 5) ? __expf(logit - mx) : 0.f;
  float den = wave_sum(e);
  float wgt = e / den;

  float acc = 0.f;
  #pragma unroll
  for (int p = 0; p < 5; ++p) {
    int kt = (bp << 3) + st + p;
    acc += __shfl(wgt, p, 32) * qkv[(size_t)kt * 576 + 384 + hc];
  }
  out[(size_t)token * CDIM + hc] = (_Float16)acc;
}

// ---------------- layout permutes, float4 granularity (48 per token)
__global__ __launch_bounds__(256) void perm_s2t(const float4* __restrict__ in,
                                                float4* __restrict__ out) {
  int tid = blockIdx.x * 256 + threadIdx.x;            // 786432 total
  int c4 = tid % 48, tt = tid / 48;                    // tt = (b*1024+p)*8+s
  int b = tt >> 13, p = (tt >> 3) & 1023, s = tt & 7;
  int src = ((b << 3) + s) * 1024 + p;
  out[(size_t)tt * 48 + c4] = in[(size_t)src * 48 + c4];
}
__global__ __launch_bounds__(256) void perm_t2s(const float4* __restrict__ in,
                                                float4* __restrict__ out) {
  int tid = blockIdx.x * 256 + threadIdx.x;
  int c4 = tid % 48, ts = tid / 48;                    // ts = (b*8+s)*1024+p
  int b = ts >> 13, s = (ts >> 10) & 7, p = ts & 1023;
  int src = ((b << 10) + p) * 8 + s;
  out[(size_t)ts * 48 + c4] = in[(size_t)src * 48 + c4];
}

extern "C" void kernel_launch(void* const* d_in, const int* in_sizes, int n_in,
                              void* d_out, int out_size, void* d_ws, size_t ws_size,
                              hipStream_t stream) {
  const float* in_x = (const float*)d_in[0];
  const float *lnG[2][2][2], *lnB[2][2][2], *qkvB[2], *rpbP[2], *projB[2], *ff1B[2], *ff2B[2];
  // stage 0 = spatial (s_*), stage 1 = temporal (t_*)
  lnG[0][0][0] = (const float*)d_in[1];  lnB[0][0][0] = (const float*)d_in[2];   // s_ln1
  lnG[0][0][1] = (const float*)d_in[8];  lnB[0][0][1] = (const float*)d_in[9];   // s_ln2
  lnG[1][0][0] = (const float*)d_in[14]; lnB[1][0][0] = (const float*)d_in[15];  // t_ln1
  lnG[1][0][1] = (const float*)d_in[21]; lnB[1][0][1] = (const float*)d_in[22];  // t_ln2
  const float* qkvW[2]  = { (const float*)d_in[3],  (const float*)d_in[16] };
  qkvB[0]  = (const float*)d_in[4];  qkvB[1]  = (const float*)d_in[17];
  rpbP[0]  = (const float*)d_in[5];  rpbP[1]  = (const float*)d_in[18];
  const float* projW[2] = { (const float*)d_in[6],  (const float*)d_in[19] };
  projB[0] = (const float*)d_in[7];  projB[1] = (const float*)d_in[20];
  const float* ff1W[2]  = { (const float*)d_in[10], (const float*)d_in[23] };
  ff1B[0]  = (const float*)d_in[11]; ff1B[1]  = (const float*)d_in[24];
  const float* ff2W[2]  = { (const float*)d_in[12], (const float*)d_in[25] };
  ff2B[0]  = (const float*)d_in[13]; ff2B[1]  = (const float*)d_in[26];

  char* ws = (char*)d_ws;
  size_t cur = 0;
  auto alloc = [&](size_t bytes) -> char* {
    char* p = ws + cur; cur += (bytes + 255) & ~(size_t)255; return p;
  };
  float*    xA   = (float*)   alloc((size_t)NTOK * CDIM * 4);
  float*    xB   = (float*)   alloc((size_t)NTOK * CDIM * 4);
  _Float16* y16  = (_Float16*)alloc((size_t)NTOK * CDIM * 2);
  float*    qkvb = (float*)   alloc((size_t)NTOK * 576 * 4);
  _Float16* a16  = (_Float16*)alloc((size_t)NTOK * CDIM * 2);
  _Float16* ff16 = (_Float16*)alloc((size_t)NTOK * MLPDIM * 2);
  _Float16 *wQ[2][2], *wP[2][2], *wF1[2][2], *wF2[2][2];
  for (int st = 0; st < 2; ++st)
    for (int li = 0; li < 2; ++li) {
      wQ [st][li] = (_Float16*)alloc((size_t)CDIM * 576 * 2);
      wP [st][li] = (_Float16*)alloc((size_t)CDIM * CDIM * 2);
      wF1[st][li] = (_Float16*)alloc((size_t)CDIM * MLPDIM * 2);
      wF2[st][li] = (_Float16*)alloc((size_t)MLPDIM * CDIM * 2);
    }

  // pack all weights to WMMA B-fragment layout (f16)
  for (int st = 0; st < 2; ++st)
    for (int li = 0; li < 2; ++li) {
      pack_w<<<(CDIM * 576) / 256, 256, 0, stream>>>(qkvW[st] + (size_t)li * CDIM * 576, wQ[st][li], CDIM, 576);
      pack_w<<<(CDIM * CDIM) / 256, 256, 0, stream>>>(projW[st] + (size_t)li * CDIM * CDIM, wP[st][li], CDIM, CDIM);
      pack_w<<<(CDIM * MLPDIM) / 256, 256, 0, stream>>>(ff1W[st] + (size_t)li * CDIM * MLPDIM, wF1[st][li], CDIM, MLPDIM);
      pack_w<<<(MLPDIM * CDIM) / 256, 256, 0, stream>>>(ff2W[st] + (size_t)li * MLPDIM * CDIM, wF2[st][li], MLPDIM, CDIM);
    }

  hipMemcpyAsync(xA, in_x, (size_t)NTOK * CDIM * 4, hipMemcpyDeviceToDevice, stream);

  // gemm grid: (N/32)*(Cout/32) waves, 8 waves per block
  const int gQKV = (NTOK / 32) * (576 / 32) / 8;    // 1152
  const int gPRJ = (NTOK / 32) * (CDIM / 32) / 8;   // 384
  const int gFF1 = (NTOK / 32) * (MLPDIM / 32) / 8; // 1536
  const int gFF2 = gPRJ;

  auto stage = [&](int st, int li, float* x) {
    const int lnBlk = NTOK / 8;
    ln_kernel<<<lnBlk, 256, 0, stream>>>(x, lnG[st][0][0] + li * CDIM, lnB[st][0][0] + li * CDIM, y16, NTOK);
    gemm16<<<gQKV, 256, 0, stream>>>(y16, wQ[st][li], qkvB[st] + li * 576, nullptr, qkvb, nullptr, NTOK, CDIM, 576, 0);
    if (st == 0)
      na2d_kernel<<<12288, 256, 0, stream>>>(qkvb, rpbP[0] + li * NHEAD * 81, a16);
    else
      na1d_kernel<<<12288, 256, 0, stream>>>(qkvb, rpbP[1] + li * NHEAD * 9, a16);
    gemm16<<<gPRJ, 256, 0, stream>>>(a16, wP[st][li], projB[st] + li * CDIM, x, x, nullptr, NTOK, CDIM, CDIM, 0);
    ln_kernel<<<lnBlk, 256, 0, stream>>>(x, lnG[st][0][1] + li * CDIM, lnB[st][0][1] + li * CDIM, y16, NTOK);
    gemm16<<<gFF1, 256, 0, stream>>>(y16, wF1[st][li], ff1B[st] + li * MLPDIM, nullptr, nullptr, ff16, NTOK, CDIM, MLPDIM, 1);
    gemm16<<<gFF2, 256, 0, stream>>>(ff16, wF2[st][li], ff2B[st] + li * CDIM, x, x, nullptr, NTOK, MLPDIM, CDIM, 0);
  };

  for (int li = 0; li < 2; ++li) {
    stage(0, li, xA);                                   // spatial
    perm_s2t<<<3072, 256, 0, stream>>>((const float4*)xA, (float4*)xB);
    stage(1, li, xB);                                   // temporal
    perm_t2s<<<3072, 256, 0, stream>>>((const float4*)xB,
                                       li == 1 ? (float4*)d_out : (float4*)xA);
  }
}
